// FFiNetTrModel_2542620639730
// MI455X (gfx1250) — matrix operand order repaired and checked by reading, not compiled, not run on hardware
//
#include <hip/hip_runtime.h>

// FFiNet 3-hop graph attention for MI455X (gfx1250, wave32).
// GEMMs on v_wmma_f32_16x16x32_bf16 (fp32 accumulate). Projection tables are
// stored bf16 (102 MB -> resident in 192 MB L2) to halve gather traffic.
// A-tiles are staged into LDS pre-swizzled into the WMMA A-fragment layout
// (bf16), so each wave loads its A operand with 2x ds_load_b128 per K-step.

#define HD 256
#define LEAKY_NEG 0.2f

typedef __attribute__((ext_vector_type(16))) __bf16         v16bf;
typedef __attribute__((ext_vector_type(16))) unsigned short v16u;
typedef __attribute__((ext_vector_type(8)))  float          v8f;

__device__ __forceinline__ unsigned short f2bf(float f) {
  unsigned u = __float_as_uint(f);
  u += 0x7fffu + ((u >> 16) & 1u);           // round-to-nearest-even
  return (unsigned short)(u >> 16);
}
__device__ __forceinline__ float bf2f(unsigned short h) {
  return __uint_as_float(((unsigned)h) << 16);
}
// monotonic float<->uint encoding for atomicMax over signed floats
__device__ __forceinline__ unsigned encf(float f) {
  unsigned u = __float_as_uint(f);
  return (u & 0x80000000u) ? ~u : (u | 0x80000000u);
}
__device__ __forceinline__ float decf(unsigned u) {
  return (u & 0x80000000u) ? __uint_as_float(u & 0x7fffffffu)
                           : __uint_as_float(~u);
}

// Stage a 16xK fp32 tile (row gr = row0+r of S, leading dim K) into LDS as
// bf16 WMMA A-fragments: dst[kt*512 + lane*16 + j], kt = k>>5,
// lane = r + ((k>>3)&1)*16, j = (k&7) + ((k&16)>>1).
__device__ __forceinline__ void stage_A_bf16(const float* __restrict__ S,
                                             int row0, int N, int K,
                                             unsigned short* __restrict__ dst,
                                             int tid) {
  int qpr = K >> 2;                          // float4 per row
  for (int i = tid; i < 16 * qpr; i += 128) {
    int r = i / qpr;
    int k4 = (i - r * qpr) << 2;             // k base, multiple of 4
    int gr = row0 + r;
    float4 v = make_float4(0.f, 0.f, 0.f, 0.f);
    if (gr < N) v = *(const float4*)(S + (size_t)gr * K + k4);
    int kt = k4 >> 5;
    int lane_t = r + (((k4 >> 3) & 1) << 4);
    int j = (k4 & 7) + ((k4 & 16) >> 1);
    ushort4 u4 = make_ushort4(f2bf(v.x), f2bf(v.y), f2bf(v.z), f2bf(v.w));
    *(ushort4*)(&dst[kt * 512 + lane_t * 16 + j]) = u4;
  }
}

// ---------------------------------------------------------------- zero
__global__ void zero_f_kernel(float* __restrict__ p, long long n) {
  long long i = (long long)blockIdx.x * blockDim.x + threadIdx.x;
  long long st = (long long)gridDim.x * blockDim.x;
  for (; i < n; i += st) p[i] = 0.f;
}

// ------------------------------------------------- sinusoidal PE table [4,256]
__global__ void pe_kernel(float* __restrict__ pe) {
  int idx = blockIdx.x * blockDim.x + threadIdx.x;
  if (idx >= 4 * HD) return;
  int p = idx >> 8, col = idx & 255;
  float i2 = (float)(col & ~1);
  float div = expf(-9.210340371976184f * i2 * (1.f / 256.f)); // ln(1e4)
  float pos = (float)p;
  pe[idx] = (col & 1) ? cosf(pos * div) : sinf(pos * div);
}

// ------------------------------------------------- pack W[K,256] -> B frags
// fragment(kt,nt): 32 lanes x 16 bf16; h=lane>>4, col=nt*16+(lane&15)
//   K=kt*32+8h+j (j<8), K=kt*32+16+8h+(j-8) (j>=8)
__global__ void pack_weights_kernel(const float* __restrict__ W,
                                    unsigned short* __restrict__ P, int K) {
  int idx = blockIdx.x * blockDim.x + threadIdx.x;
  int total = (K >> 5) * 16 * 512;
  if (idx >= total) return;
  int j = idx & 15;
  int lane = (idx >> 4) & 31;
  int frag = idx >> 9;
  int nt = frag & 15;
  int kt = frag >> 4;
  int k = (kt << 5) + ((j & 8) << 1) + (((lane >> 4) & 1) << 3) + (j & 7);
  int c = (nt << 4) + (lane & 15);
  P[idx] = f2bf(W[(size_t)k * HD + c]);
}

// ------------------------------------------ 4 projections: x@W + PE (bf16 out)
__global__ __launch_bounds__(128)
void proj_gemm_kernel(const float* __restrict__ x,
                      const unsigned short* __restrict__ Wpack,
                      const float* __restrict__ peBuf,
                      unsigned short* __restrict__ proj, int N) {
  __shared__ unsigned short xsb[4 * 512];    // A fragments, bf16
  int tid = threadIdx.x, wave = tid >> 5, lane = tid & 31;
  int p = blockIdx.y;                        // 0:src 1:mid2 2:mid1 3:dst
  int row0 = blockIdx.x * 16;
  const unsigned short* Wp = Wpack + (size_t)p * (4 * 16 * 512);
  unsigned short* out = proj + (size_t)p * N * HD;

  stage_A_bf16(x, row0, N, 128, xsb, tid);
  __syncthreads();

  v8f acc[4] = {};
  for (int kt = 0; kt < 4; ++kt) {
    v16bf a = __builtin_bit_cast(v16bf, *(const v16u*)(&xsb[kt * 512 + lane * 16]));
#pragma unroll
    for (int nt = 0; nt < 4; ++nt) {
      int ntile = wave * 4 + nt;
      v16bf b = __builtin_bit_cast(
          v16bf, *(const v16u*)(Wp + ((size_t)(kt * 16 + ntile) * 512 + lane * 16)));
      acc[nt] = __builtin_amdgcn_wmma_f32_16x16x32_bf16(
          false, a, false, b, (short)0, acc[nt], false, false);
    }
  }

#pragma unroll
  for (int nt = 0; nt < 4; ++nt) {
    int col = (wave * 4 + nt) * 16 + (lane & 15);
    float pev = peBuf[p * HD + col];
#pragma unroll
    for (int r8 = 0; r8 < 8; ++r8) {
      int row = ((lane >> 4) << 3) + r8;     // C layout: lanes>=16 hold M=8..15
      int grow = row0 + row;
      if (grow < N) out[(size_t)grow * HD + col] = f2bf(acc[nt][r8] + pev);
    }
  }
}

// ----------------------------------------------------- hop pass A: logits+max
__global__ __launch_bounds__(256)
void hop_logits_kernel(int nItems, int nGather,
    const int* __restrict__ i0, const int* __restrict__ i1,
    const int* __restrict__ i2, const int* __restrict__ i3,
    const unsigned short* __restrict__ proj, int p0, int p1, int p2, int p3,
    int N,
    const float* __restrict__ g1, int r1, const float* __restrict__ Wg1,
    const float* __restrict__ bg1,
    const float* __restrict__ g2, int r2, const float* __restrict__ Wg2,
    const float* __restrict__ bg2,
    const float* __restrict__ attnW,
    float* __restrict__ logits, unsigned* __restrict__ gmaxEnc) {
  __shared__ float smax[256];
  int tid = threadIdx.x;
  int item = blockIdx.x * 8 + (tid >> 5);
  int lane = tid & 31;
  bool valid = item < nItems;
  float partial = 0.f;
  if (valid) {
    size_t NN = (size_t)N * HD;
    const unsigned short* P0 = proj + (size_t)p0 * NN + (size_t)i0[item] * HD;
    const unsigned short* P1 = proj + (size_t)p1 * NN + (size_t)i1[item] * HD;
    const unsigned short* P2 =
        (nGather > 2) ? proj + (size_t)p2 * NN + (size_t)i2[item] * HD : nullptr;
    const unsigned short* P3 =
        (nGather > 3) ? proj + (size_t)p3 * NN + (size_t)i3[item] * HD : nullptr;
    float c1[6], c2[3];
    for (int r = 0; r < r1; ++r) c1[r] = g1[(size_t)item * r1 + r];
    for (int r = 0; r < r2; ++r) c2[r] = g2[(size_t)item * r2 + r];
#pragma unroll
    for (int u = 0; u < 8; ++u) {
      int c = lane * 8 + u;                  // head = c>>5 = lane>>2
      float sg = bf2f(P0[c]) + bf2f(P1[c]);
      if (nGather > 2) sg += bf2f(P2[c]);
      if (nGather > 3) sg += bf2f(P3[c]);
      float m = bg1[c];
      for (int r = 0; r < r1; ++r) m += c1[r] * Wg1[r * HD + c];
      if (r2 > 0) {
        m += bg2[c];
        for (int r = 0; r < r2; ++r) m += c2[r] * Wg2[r * HD + c];
      }
      float v = sg * m;
      v = (v >= 0.f) ? v : LEAKY_NEG * v;
      partial += attnW[c] * v;
    }
  }
  // reduce across the 4 lanes of each head group
  partial += __shfl_xor(partial, 1, 32);
  partial += __shfl_xor(partial, 2, 32);
  if (valid && (lane & 3) == 0)
    logits[(size_t)item * 8 + (lane >> 2)] = partial;

  smax[tid] = valid ? partial : -1e30f;
  __syncthreads();
  for (int s = 128; s > 0; s >>= 1) {
    if (tid < s) smax[tid] = fmaxf(smax[tid], smax[tid + s]);
    __syncthreads();
  }
  if (tid == 0) atomicMax(gmaxEnc, encf(smax[0]));
}

// ------------------------------------------------ hop pass B: exp + segsum
__global__ void hop_expsum_kernel(int nItems, float* __restrict__ ab,
                                  const int* __restrict__ dstIdx,
                                  float* __restrict__ s,
                                  const unsigned* __restrict__ gmaxEnc) {
  int idx = blockIdx.x * blockDim.x + threadIdx.x;
  if (idx >= nItems * 8) return;
  float gmax = decf(*gmaxEnc);
  int item = idx >> 3, h = idx & 7;
  float a = expf(ab[idx] - gmax);
  ab[idx] = a;
  atomicAdd(&s[(size_t)dstIdx[item] * 8 + h], a);
}

// ------------------------------------------------ hop pass C: weighted scatter
__global__ __launch_bounds__(256)
void hop_scatter_kernel(int nItems, const float* __restrict__ aBuf,
                        const int* __restrict__ srcIdx,
                        const int* __restrict__ dstIdx,
                        const unsigned short* __restrict__ projVal,
                        const float* __restrict__ s,
                        float* __restrict__ outAcc) {
  int tid = threadIdx.x;
  int item = blockIdx.x * 8 + (tid >> 5);
  if (item >= nItems) return;
  int lane = tid & 31;
  int si = srcIdx[item], di = dstIdx[item];
  int head = lane >> 2;
  float alpha = aBuf[(size_t)item * 8 + head] / (s[(size_t)di * 8 + head] + 1e-16f);
  const unsigned short* V = projVal + (size_t)si * HD;
  float* O = outAcc + (size_t)di * HD;
#pragma unroll
  for (int u = 0; u < 8; ++u) {
    int c = lane * 8 + u;
    atomicAdd(&O[c], bf2f(V[c]) * alpha);
  }
}

// ------------------------ final: 3 GEMMs + residual GEMM + bias + LN + PReLU
__global__ __launch_bounds__(128)
void final_gemm_ln_kernel(
    const float* __restrict__ eo, const float* __restrict__ ao,
    const float* __restrict__ dho, const float* __restrict__ x,
    const unsigned short* __restrict__ P1, const unsigned short* __restrict__ P2,
    const unsigned short* __restrict__ P3, const unsigned short* __restrict__ Pr,
    const float* __restrict__ b1, const float* __restrict__ b2,
    const float* __restrict__ b3, const float* __restrict__ bias,
    const float* __restrict__ lng, const float* __restrict__ lnb,
    const float* __restrict__ pw, float* __restrict__ out, int N) {
  __shared__ unsigned short Asb[8 * 512];    // A fragments, bf16
  __shared__ float rsum[16], rsq[16];
  int tid = threadIdx.x, wave = tid >> 5, lane = tid & 31;
  int row0 = blockIdx.x * 16;
  const float* srcs[4] = {eo, ao, dho, x};
  const unsigned short* Ws[4] = {P1, P2, P3, Pr};
  const int Ks[4] = {256, 256, 256, 128};
  v8f acc[4] = {};

  for (int m = 0; m < 4; ++m) {
    int K = Ks[m];
    __syncthreads();
    stage_A_bf16(srcs[m], row0, N, K, Asb, tid);
    __syncthreads();
    int nKt = K >> 5;
    for (int kt = 0; kt < nKt; ++kt) {
      v16bf a = __builtin_bit_cast(v16bf, *(const v16u*)(&Asb[kt * 512 + lane * 16]));
#pragma unroll
      for (int nt = 0; nt < 4; ++nt) {
        int ntile = wave * 4 + nt;
        v16bf b = __builtin_bit_cast(
            v16bf,
            *(const v16u*)(Ws[m] + ((size_t)(kt * 16 + ntile) * 512 + lane * 16)));
        acc[nt] = __builtin_amdgcn_wmma_f32_16x16x32_bf16(
            false, a, false, b, (short)0, acc[nt], false, false);
      }
    }
  }

  if (tid < 16) { rsum[tid] = 0.f; rsq[tid] = 0.f; }
  __syncthreads();
  float vv[4][8];
#pragma unroll
  for (int r8 = 0; r8 < 8; ++r8) {
    float s1 = 0.f, s2 = 0.f;
#pragma unroll
    for (int nt = 0; nt < 4; ++nt) {
      int col = (wave * 4 + nt) * 16 + (lane & 15);
      float v = acc[nt][r8] + b1[col] + b2[col] + b3[col] + bias[col];
      vv[nt][r8] = v;
      s1 += v;
      s2 += v * v;
    }
    int row = ((lane >> 4) << 3) + r8;
    atomicAdd(&rsum[row], s1);
    atomicAdd(&rsq[row], s2);
  }
  __syncthreads();
  float slope = pw[0];
#pragma unroll
  for (int r8 = 0; r8 < 8; ++r8) {
    int row = ((lane >> 4) << 3) + r8;
    int grow = row0 + row;
    if (grow >= N) continue;
    float mean = rsum[row] * (1.f / 256.f);
    float var = rsq[row] * (1.f / 256.f) - mean * mean;
    float inv = rsqrtf(var + 1e-5f);
#pragma unroll
    for (int nt = 0; nt < 4; ++nt) {
      int col = (wave * 4 + nt) * 16 + (lane & 15);
      float y = (vv[nt][r8] - mean) * inv * lng[col] + lnb[col];
      y = (y >= 0.f) ? y : slope * y;
      out[(size_t)grow * HD + col] = y;
    }
  }
}

// ===========================================================================
extern "C" void kernel_launch(void* const* d_in, const int* in_sizes, int n_in,
                              void* d_out, int out_size, void* d_ws,
                              size_t ws_size, hipStream_t stream) {
  const float* x    = (const float*)d_in[0];
  const int* e_src  = (const int*)d_in[1];
  const int* e_dst  = (const int*)d_in[2];
  const int* a_src  = (const int*)d_in[3];
  const int* a_mid  = (const int*)d_in[4];
  const int* a_dst  = (const int*)d_in[5];
  const int* q_src  = (const int*)d_in[6];
  const int* q_mid2 = (const int*)d_in[7];
  const int* q_mid1 = (const int*)d_in[8];
  const int* q_dst  = (const int*)d_in[9];
  const float* dist1 = (const float*)d_in[10];
  const float* dista = (const float*)d_in[11];
  const float* angm  = (const float*)d_in[12];
  const float* distd = (const float*)d_in[13];
  const float* dihm  = (const float*)d_in[14];
  const float* W_src  = (const float*)d_in[15];
  const float* W_dst  = (const float*)d_in[16];
  const float* W_mid1 = (const float*)d_in[17];
  const float* W_mid2 = (const float*)d_in[18];
  const float* W_pb  = (const float*)d_in[19];
  const float* b_pb  = (const float*)d_in[20];
  const float* W_pu1 = (const float*)d_in[21];
  const float* b_pu1 = (const float*)d_in[22];
  const float* W_pu2 = (const float*)d_in[23];
  const float* b_pu2 = (const float*)d_in[24];
  const float* W_ang = (const float*)d_in[25];
  const float* b_ang = (const float*)d_in[26];
  const float* W_dih = (const float*)d_in[27];
  const float* b_dih = (const float*)d_in[28];
  const float* W_one   = (const float*)d_in[29];
  const float* b_one   = (const float*)d_in[30];
  const float* W_two   = (const float*)d_in[31];
  const float* b_two   = (const float*)d_in[32];
  const float* W_three = (const float*)d_in[33];
  const float* b_three = (const float*)d_in[34];
  const float* dattn = (const float*)d_in[35];
  const float* tattn = (const float*)d_in[36];
  const float* qattn = (const float*)d_in[37];
  const float* bias  = (const float*)d_in[38];
  const float* W_res = (const float*)d_in[39];
  const float* ln_g  = (const float*)d_in[40];
  const float* ln_b  = (const float*)d_in[41];
  const float* prelu = (const float*)d_in[42];

  const int N = in_sizes[0] / 128;
  const int E = in_sizes[1];
  const int T = in_sizes[3];
  const int Q = in_sizes[6];
  int mx = E > T ? E : T; if (Q > mx) mx = Q;

  char* ws = (char*)d_ws;
  size_t off = 0;
  auto alloc = [&](size_t bytes) -> void* {
    void* p = ws + off;
    off = (off + bytes + 255) & ~(size_t)255;
    return p;
  };
  unsigned short* Ppj = (unsigned short*)alloc((size_t)4 * 32768 * 2);
  unsigned short* P1w = (unsigned short*)alloc((size_t)65536 * 2);
  unsigned short* P2w = (unsigned short*)alloc((size_t)65536 * 2);
  unsigned short* P3w = (unsigned short*)alloc((size_t)65536 * 2);
  unsigned short* Prw = (unsigned short*)alloc((size_t)32768 * 2);
  float* peBuf = (float*)alloc((size_t)4 * HD * 4);
  unsigned short* proj = (unsigned short*)alloc((size_t)4 * N * HD * 2);
  size_t zstart = off;
  float* hop0 = (float*)alloc((size_t)N * HD * 4);
  float* hop1 = (float*)alloc((size_t)N * HD * 4);
  float* hop2 = (float*)alloc((size_t)N * HD * 4);
  float* s0 = (float*)alloc((size_t)N * 8 * 4);
  float* s1 = (float*)alloc((size_t)N * 8 * 4);
  float* s2 = (float*)alloc((size_t)N * 8 * 4);
  unsigned* gmax = (unsigned*)alloc(3 * 4);
  size_t zend = off;
  float* logitsBuf = (float*)alloc((size_t)mx * 8 * 4);
  (void)ws_size; (void)n_in; (void)out_size;

  // zero accumulators + softmax sums + encoded maxima (enc is >=0 monotonic)
  zero_f_kernel<<<4096, 256, 0, stream>>>((float*)(ws + zstart),
                                          (long long)((zend - zstart) / 4));

  pe_kernel<<<4, 256, 0, stream>>>(peBuf);

  // pack weights into WMMA B-fragment layout (bf16)
  pack_weights_kernel<<<(128 * 256 + 255) / 256, 256, 0, stream>>>(W_src,  Ppj + 0 * 32768, 128);
  pack_weights_kernel<<<(128 * 256 + 255) / 256, 256, 0, stream>>>(W_mid2, Ppj + 1 * 32768, 128);
  pack_weights_kernel<<<(128 * 256 + 255) / 256, 256, 0, stream>>>(W_mid1, Ppj + 2 * 32768, 128);
  pack_weights_kernel<<<(128 * 256 + 255) / 256, 256, 0, stream>>>(W_dst,  Ppj + 3 * 32768, 128);
  pack_weights_kernel<<<(256 * 256 + 255) / 256, 256, 0, stream>>>(W_one,   P1w, 256);
  pack_weights_kernel<<<(256 * 256 + 255) / 256, 256, 0, stream>>>(W_two,   P2w, 256);
  pack_weights_kernel<<<(256 * 256 + 255) / 256, 256, 0, stream>>>(W_three, P3w, 256);
  pack_weights_kernel<<<(128 * 256 + 255) / 256, 256, 0, stream>>>(W_res,   Prw, 128);

  // projections: p=0 src(pe0), 1 mid2(pe1), 2 mid1(pe2), 3 dst(pe3)
  dim3 pg((N + 15) / 16, 4);
  proj_gemm_kernel<<<pg, 128, 0, stream>>>(x, Ppj, peBuf, proj, N);

  // ---- hop 0: bonded edges (mid1[src]+dst[dst]) * (dist@W_pb+b_pb)
  hop_logits_kernel<<<(E + 7) / 8, 256, 0, stream>>>(E, 2,
      e_src, e_dst, e_src, e_src, proj, 2, 3, 0, 0, N,
      dist1, 2, W_pb, b_pb, nullptr, 0, nullptr, nullptr,
      dattn, logitsBuf, gmax + 0);
  hop_expsum_kernel<<<(E * 8 + 255) / 256, 256, 0, stream>>>(E, logitsBuf, e_dst, s0, gmax + 0);
  hop_scatter_kernel<<<(E + 7) / 8, 256, 0, stream>>>(E, logitsBuf, e_src, e_dst,
      proj + (size_t)2 * N * HD, s0, hop0);

  // ---- hop 1: angles (mid2[s]+dst[d]+mid1[m]) * (ang@W_ang+b + dist@W_pu1+b)
  hop_logits_kernel<<<(T + 7) / 8, 256, 0, stream>>>(T, 3,
      a_src, a_dst, a_mid, a_src, proj, 1, 3, 2, 0, N,
      angm, 2, W_ang, b_ang, dista, 3, W_pu1, b_pu1,
      tattn, logitsBuf, gmax + 1);
  hop_expsum_kernel<<<(T * 8 + 255) / 256, 256, 0, stream>>>(T, logitsBuf, a_dst, s1, gmax + 1);
  hop_scatter_kernel<<<(T + 7) / 8, 256, 0, stream>>>(T, logitsBuf, a_src, a_dst,
      proj + (size_t)1 * N * HD, s1, hop1);

  // ---- hop 2: dihedrals (src+dst+mid1+mid2) * (dih@W_dih+b + dist@W_pu2+b)
  hop_logits_kernel<<<(Q + 7) / 8, 256, 0, stream>>>(Q, 4,
      q_src, q_dst, q_mid1, q_mid2, proj, 0, 3, 2, 1, N,
      dihm, 6, W_dih, b_dih, distd, 3, W_pu2, b_pu2,
      qattn, logitsBuf, gmax + 2);
  hop_expsum_kernel<<<(Q * 8 + 255) / 256, 256, 0, stream>>>(Q, logitsBuf, q_dst, s2, gmax + 2);
  hop_scatter_kernel<<<(Q + 7) / 8, 256, 0, stream>>>(Q, logitsBuf, q_src, q_dst,
      proj + (size_t)0 * N * HD, s2, hop2);

  // ---- final fused GEMMs + LayerNorm + PReLU
  final_gemm_ln_kernel<<<(N + 15) / 16, 128, 0, stream>>>(hop0, hop1, hop2, x,
      P1w, P2w, P3w, Prw, b_one, b_two, b_three, bias, ln_g, ln_b, prelu,
      (float*)d_out, N);
}
